// S4Net_33947421507700
// MI455X (gfx1250) — matrix operand order repaired and checked by reading, not compiled, not run on hardware
//
#include <hip/hip_runtime.h>
#include <hip/hip_bf16.h>
#include <math.h>

// Problem dims (compile-time constants from the reference)
constexpr int kNL = 4;
constexpr int kH  = 1024;   // model dim
constexpr int kN  = 32;     // complex states per channel
constexpr int kB  = 8;      // batch
constexpr int kL  = 1024;   // seq length

typedef __attribute__((ext_vector_type(16))) __bf16 v16bf;
typedef __attribute__((ext_vector_type(8)))  float  v8f;
typedef unsigned int v4u __attribute__((ext_vector_type(4)));
typedef int          v4i __attribute__((ext_vector_type(4)));
typedef int          v8i __attribute__((ext_vector_type(8)));

// ---- CDNA5 feature probes (device pass only; independent guards!) ----
#if defined(__gfx1250__)
#  if __has_builtin(__builtin_amdgcn_tensor_load_to_lds) && \
      __has_builtin(__builtin_amdgcn_s_wait_tensorcnt)
#    define S4_HAVE_TDM 1
#  endif
#  if __has_builtin(__builtin_amdgcn_global_load_async_to_lds_b128) && \
      __has_builtin(__builtin_amdgcn_s_wait_asynccnt)
#    define S4_HAVE_ASYNC 1
#  endif
#endif

struct Q2 { uint4 lo; uint4 hi; };

static __device__ __forceinline__ v16bf make_frag(uint4 lo, uint4 hi) {
  Q2 q{lo, hi};
  return __builtin_bit_cast(v16bf, q);
}

// float -> bf16 round-to-nearest-even (raw bits)
static __device__ __forceinline__ unsigned short f2bf(float f) {
  unsigned int u = __builtin_bit_cast(unsigned int, f);
  u += 0x7FFFu + ((u >> 16) & 1u);
  return (unsigned short)(u >> 16);
}

#if S4_HAVE_TDM
// Issue one 2D TDM descriptor: tile of (tileCols x tileRows) 2-byte elements,
// row stride srcStride elements, global tile start gsrc -> LDS byte offset.
// (ISA ch.8 D# groups 0/1; groups 2/3 zero for 2D.)
static __device__ __forceinline__ void s4_tdm_load_tile(
    const unsigned short* gsrc, unsigned ldsoff,
    int tileCols, int tileRows, int srcStride) {
  const unsigned long long ga = (unsigned long long)(uintptr_t)gsrc;
  v4u g0;
  g0.x = 1u;                                    // count=1, user descriptor
  g0.y = ldsoff;                                // lds_addr (bytes)
  g0.z = (unsigned)(ga & 0xFFFFFFFFull);        // global_addr[31:0]
  g0.w = ((unsigned)(ga >> 32) & 0x01FFFFFFu)   // global_addr[56:32]
         | (2u << 30);                          // type=2 ("image")
  v8i g1;
  g1[0] = (int)(1u << 16);                      // data_size=1 -> 2 bytes
  g1[1] = (int)((unsigned)tileCols << 16);      // tensor_dim0 (lo16)
  g1[2] = (int)((unsigned)tileRows << 16);      // tensor_dim1 (lo16)
  g1[3] = (int)((unsigned)tileCols << 16);      // tile_dim0
  g1[4] = tileRows;                             // tile_dim1
  g1[5] = srcStride;                            // tensor_dim0_stride (lo32)
  g1[6] = 0;
  g1[7] = 0;
  v4i gz = {0, 0, 0, 0};
#if __clang_major__ >= 23
  v8i gz8 = {0, 0, 0, 0, 0, 0, 0, 0};
  __builtin_amdgcn_tensor_load_to_lds(g0, g1, gz, gz, gz8, 0);
#else
  __builtin_amdgcn_tensor_load_to_lds(g0, g1, gz, gz, 0);
#endif
}
#endif

// ---------------------------------------------------------------------------
// (B,L,H) -> (B,H,L) tiled transpose
// ---------------------------------------------------------------------------
__global__ __launch_bounds__(256) void s4_transpose_in(
    const float* __restrict__ x, float* __restrict__ hdst) {
  __shared__ float tile[32][33];
  const int b  = blockIdx.z;
  const int l0 = blockIdx.x * 32;
  const int h0 = blockIdx.y * 32;
  const int tx = threadIdx.x & 31;
  const int ty = threadIdx.x >> 5;      // 0..7
  #pragma unroll
  for (int i = 0; i < 32; i += 8)
    tile[ty + i][tx] = x[((size_t)b * kL + l0 + ty + i) * kH + h0 + tx];
  __syncthreads();
  #pragma unroll
  for (int i = 0; i < 32; i += 8)
    hdst[((size_t)b * kH + h0 + ty + i) * kL + l0 + tx] = tile[tx][ty + i];
}

// (B,H,L) -> (B,L,H)
__global__ __launch_bounds__(256) void s4_transpose_out(
    const float* __restrict__ hsrc, float* __restrict__ out) {
  __shared__ float tile[32][33];
  const int b  = blockIdx.z;
  const int h0 = blockIdx.x * 32;
  const int l0 = blockIdx.y * 32;
  const int tx = threadIdx.x & 31;
  const int ty = threadIdx.x >> 5;
  #pragma unroll
  for (int i = 0; i < 32; i += 8)
    tile[ty + i][tx] = hsrc[((size_t)b * kH + h0 + ty + i) * kL + l0 + tx];
  __syncthreads();
  #pragma unroll
  for (int i = 0; i < 32; i += 8)
    out[((size_t)b * kL + l0 + ty + i) * kH + h0 + tx] = tile[tx][ty + i];
}

// ---------------------------------------------------------------------------
// Per-layer constants: lambda = exp(dt*A), C' = 2*C*(lambda-1)/A  (complex)
// ---------------------------------------------------------------------------
__global__ __launch_bounds__(256) void s4_prep_consts(
    const float* __restrict__ log_dt, const float* __restrict__ A_re_log,
    const float* __restrict__ A_im,   const float* __restrict__ C_re,
    const float* __restrict__ C_im,
    float* __restrict__ lamr, float* __restrict__ lami,
    float* __restrict__ ccr,  float* __restrict__ cci) {
  const int idx = blockIdx.x * 256 + threadIdx.x;
  if (idx >= kH * kN) return;
  const int hh = idx >> 5;              // idx / kN
  const float Are = -__expf(A_re_log[idx]);
  const float Aim = A_im[idx];
  const float dt  = __expf(log_dt[hh]);
  const float dr  = Are * dt, di = Aim * dt;
  const float e   = __expf(dr);
  const float lr  = e * __cosf(di);
  const float li  = e * __sinf(di);
  const float er = lr - 1.f, ei = li;   // expm1(dtA) = lambda - 1
  const float inv = 1.f / (Are * Are + Aim * Aim);
  const float fr = (er * Are + ei * Aim) * inv;   // (lambda-1)/A
  const float fi = (ei * Are - er * Aim) * inv;
  const float c0r = C_re[idx], c0i = C_im[idx];
  lamr[idx] = lr;
  lami[idx] = li;
  ccr[idx]  = 2.f * (c0r * fr - c0i * fi);        // fold the "2*Re" factor
  cci[idx]  = 2.f * (c0r * fi + c0i * fr);
}

__global__ __launch_bounds__(256) void s4_prep_w(
    const float* __restrict__ Wsrc, unsigned short* __restrict__ Wb, int n) {
  const int i = blockIdx.x * 256 + threadIdx.x;
  if (i < n) Wb[i] = f2bf(Wsrc[i]);
}

// ---------------------------------------------------------------------------
// Channel LayerNorm: h (B,H,L) -> z (B,H,L)
// ---------------------------------------------------------------------------
__global__ __launch_bounds__(256) void s4_ln_kernel(
    const float* __restrict__ hsrc, const float* __restrict__ w,
    const float* __restrict__ bvec, float* __restrict__ z) {
  __shared__ float ssum[8][32];
  __shared__ float ssq[8][32];
  __shared__ float smu[32], srs[32];
  const int lt = threadIdx.x & 31;
  const int hg = threadIdx.x >> 5;      // 0..7
  const int b  = blockIdx.x >> 5;       // kL/32 = 32 slabs per batch
  const int l0 = (blockIdx.x & 31) * 32;
  const float* base = hsrc + ((size_t)b * kH) * kL + l0 + lt;
  float s = 0.f, sq = 0.f;
  for (int hh = hg * 128; hh < hg * 128 + 128; ++hh) {
    const float v = base[(size_t)hh * kL];
    s += v; sq += v * v;
  }
  ssum[hg][lt] = s; ssq[hg][lt] = sq;
  __syncthreads();
  if (hg == 0) {
    float ts = 0.f, tq = 0.f;
    #pragma unroll
    for (int g = 0; g < 8; ++g) { ts += ssum[g][lt]; tq += ssq[g][lt]; }
    const float m = ts * (1.f / kH);
    const float var = tq * (1.f / kH) - m * m;
    smu[lt] = m;
    srs[lt] = rsqrtf(var + 1e-5f);
  }
  __syncthreads();
  const float m = smu[lt], r = srs[lt];
  float* zbase = z + ((size_t)b * kH) * kL + l0 + lt;
  for (int hh = hg * 128; hh < hg * 128 + 128; ++hh) {
    const float v = base[(size_t)hh * kL];
    zbase[(size_t)hh * kL] = (v - m) * r * w[hh] + bvec[hh];
  }
}

// ---------------------------------------------------------------------------
// S4D diagonal recurrence + D-skip + GELU, emit bf16 activations.
// One wave32 per (b,h); lane = state index n.
// ---------------------------------------------------------------------------
__global__ __launch_bounds__(256) void s4_scan_kernel(
    const float* __restrict__ z,
    const float* __restrict__ lamr, const float* __restrict__ lami,
    const float* __restrict__ ccr,  const float* __restrict__ cci,
    const float* __restrict__ Dskip, unsigned short* __restrict__ g) {
  const int gw   = blockIdx.x * 8 + (threadIdx.x >> 5);   // wave id over B*H
  const int lane = threadIdx.x & 31;
  const int b  = gw >> 10;              // gw / kH
  const int hh = gw & (kH - 1);
  const int ci = hh * kN + lane;
  const float lr = lamr[ci], li = lami[ci];
  const float cr = ccr[ci],  cim = cci[ci];
  const float D  = Dskip[hh];
  const float* zb = z + ((size_t)b * kH + hh) * kL;
  unsigned short* gb = g + ((size_t)b * kH + hh) * kL;
  float sr = 0.f, si = 0.f;
  for (int l0 = 0; l0 < kL; l0 += 32) {
    const float zv = zb[l0 + lane];
    float yk = 0.f;
    #pragma unroll
    for (int t = 0; t < 32; ++t) {
      const float zt = __shfl(zv, t, 32);
      const float nr = fmaf(lr, sr, fmaf(-li, si, zt));
      const float ni = fmaf(lr, si, li * sr);
      sr = nr; si = ni;
      float p = fmaf(cr, sr, -cim * si);
      #pragma unroll
      for (int off = 16; off > 0; off >>= 1) p += __shfl_xor(p, off, 32);
      if (lane == t) yk = p;
    }
    const float y = fmaf(D, zv, yk);
    const float u  = 0.7978845608028654f * fmaf(0.044715f, y * y * y, y);
    const float gl = 0.5f * y * (1.f + tanhf(u));
    gb[l0 + lane] = f2bf(gl);
  }
}

// ---------------------------------------------------------------------------
// Fused GEMM + GLU + residual.
//   val[o,c]  = b[o]   + sum_h W[o,   h] * g[h,c]
//   gate[o,c] = b[o+H] + sum_h W[o+H, h] * g[h,c]
//   hres[b,o,l] += val * sigmoid(gate)
// 8 waves per WG, 128x128 tile, each wave 64x32 (4x2 C-frags, x2 val/gate).
// A (weight) tiles staged by the Tensor Data Mover; raw B tile staged by
// async global->LDS loads, then transposed in LDS for B-fragment reads.
// ---------------------------------------------------------------------------
constexpr int kBTP = 40;   // LDS B-tile pitch in bf16 elems (80B, 16B aligned)

__global__ __launch_bounds__(256) void s4_gemm_glu_kernel(
    const unsigned short* __restrict__ Wb,   // (2H, H) bf16 row-major
    const unsigned short* __restrict__ G,    // (B,H,L) bf16
    const float* __restrict__ bias,          // (2H,)
    float* __restrict__ Hres) {              // (B,H,L) f32, accumulated
  __shared__ __align__(16) unsigned short Bt[128 * kBTP];   // transposed [N][K]
#if S4_HAVE_TDM
  __shared__ __align__(16) unsigned short Aval[128 * 32];   // W rows  m0..m0+127
  __shared__ __align__(16) unsigned short Agat[128 * 32];   // W rows +H
#endif
#if S4_HAVE_ASYNC
  __shared__ __align__(16) unsigned short Braw[32 * 128];   // raw row-major
#endif

  const int tid   = threadIdx.x;
  const int lane  = tid & 31;
  const int wid   = tid >> 5;
  const int waveM = wid & 1;   // 2 waves along M
  const int waveN = wid >> 1;  // 4 waves along N
  const int c0 = blockIdx.x * 128;
  const int bb = c0 >> 10;                 // batch (L=1024)
  const int l0 = c0 & (kL - 1);
  const int m0 = blockIdx.y * 128;
  const unsigned short* Gb = G + (size_t)bb * kH * kL;

  v8f accV[4][2] = {};
  v8f accG[4][2] = {};

  // per-lane fragment geometry (ISA 7.12.2, 16-bit A / B layouts)
  const int rowA   = lane & 15;
  const int khalfA = (lane >> 4) * 8;
  const int colB   = lane & 15;
  const int khalfB = (lane >> 4) * 8;

  const int kkS = tid >> 3;          // staging: K row
  const int lsS = (tid & 7) * 16;    // staging: 16-col strip

  for (int k0 = 0; k0 < kH; k0 += 32) {
    // ===== stage A tiles via Tensor Data Mover (wave 0 only) =====
#if S4_HAVE_TDM
    if (wid == 0) {
      const unsigned aoffV = (unsigned)(uintptr_t)(
          (__attribute__((address_space(3))) unsigned short*)Aval);
      const unsigned aoffG = (unsigned)(uintptr_t)(
          (__attribute__((address_space(3))) unsigned short*)Agat);
      s4_tdm_load_tile(Wb + (size_t)m0 * kH + k0,        aoffV, 32, 128, kH);
      s4_tdm_load_tile(Wb + (size_t)(m0 + kH) * kH + k0, aoffG, 32, 128, kH);
      __builtin_amdgcn_s_wait_tensorcnt(0);
    }
#endif

    // ===== stage B tile (32 x 128) and transpose into Bt =====
#if S4_HAVE_ASYNC
    {
      const unsigned short* gsrc = Gb + (size_t)(k0 + kkS) * kL + l0 + lsS;
      // param 1 per probed prototype: v4i in AS(1) ("__device__")
      __attribute__((address_space(1))) v4i* gp =
          (__attribute__((address_space(1))) v4i*)(uintptr_t)gsrc;
      // LDS destination: cast AS first (same pointee), then pointee type
      __attribute__((address_space(3))) unsigned short* lp0 =
          (__attribute__((address_space(3))) unsigned short*)&Braw[kkS * 128 + lsS];
      __attribute__((address_space(3))) v4i* lp =
          (__attribute__((address_space(3))) v4i*)lp0;
      __builtin_amdgcn_global_load_async_to_lds_b128(gp, lp, 0, 0);
      __builtin_amdgcn_global_load_async_to_lds_b128(gp, lp, 16, 0);
      __builtin_amdgcn_s_wait_asynccnt(0);
      // read back own strip and scatter transposed
      const uint4* rp = (const uint4*)&Braw[kkS * 128 + lsS];
      uint4 d0 = rp[0];
      uint4 d1 = rp[1];
      const unsigned short* u0 = (const unsigned short*)&d0;
      const unsigned short* u1 = (const unsigned short*)&d1;
      #pragma unroll
      for (int j = 0; j < 8; ++j) Bt[(lsS + j) * kBTP + kkS] = u0[j];
      #pragma unroll
      for (int j = 0; j < 8; ++j) Bt[(lsS + 8 + j) * kBTP + kkS] = u1[j];
    }
#else
    {
      const uint4* src = (const uint4*)(Gb + (size_t)(k0 + kkS) * kL + l0 + lsS);
      uint4 d0 = src[0];
      uint4 d1 = src[1];
      if (k0 + 32 < kH)
        __builtin_prefetch(Gb + (size_t)(k0 + 32 + kkS) * kL + l0 + lsS, 0, 3);
      const unsigned short* u0 = (const unsigned short*)&d0;
      const unsigned short* u1 = (const unsigned short*)&d1;
      #pragma unroll
      for (int j = 0; j < 8; ++j) Bt[(lsS + j) * kBTP + kkS] = u0[j];
      #pragma unroll
      for (int j = 0; j < 8; ++j) Bt[(lsS + 8 + j) * kBTP + kkS] = u1[j];
    }
#endif
    __syncthreads();

    // ===== B fragments (ds_load_b128 pairs) =====
    v16bf bfr[2];
    #pragma unroll
    for (int nf = 0; nf < 2; ++nf) {
      const int r = waveN * 32 + nf * 16 + colB;
      const uint4* p = (const uint4*)&Bt[r * kBTP + khalfB];
      bfr[nf] = make_frag(p[0], p[2]);
    }

    // ===== A fragments + WMMA =====
    #pragma unroll
    for (int mf = 0; mf < 4; ++mf) {
#if S4_HAVE_TDM
      const int rloc = waveM * 64 + mf * 16 + rowA;
      const uint4* pv = (const uint4*)&Aval[rloc * 32 + khalfA];
      const uint4* pg = (const uint4*)&Agat[rloc * 32 + khalfA];
      const v16bf av = make_frag(pv[0], pv[2]);   // +0B and +32B chunks
      const v16bf ag = make_frag(pg[0], pg[2]);
#else
      const int orow = m0 + waveM * 64 + mf * 16 + rowA;
      const unsigned short* aV = Wb + (size_t)orow * kH + k0 + khalfA;
      const unsigned short* aG = Wb + (size_t)(orow + kH) * kH + k0 + khalfA;
      const v16bf av = make_frag(*(const uint4*)aV, *(const uint4*)(aV + 16));
      const v16bf ag = make_frag(*(const uint4*)aG, *(const uint4*)(aG + 16));
#endif
      #pragma unroll
      for (int nf = 0; nf < 2; ++nf) {
        accV[mf][nf] = __builtin_amdgcn_wmma_f32_16x16x32_bf16(
            false, av, false, bfr[nf], (short)0, accV[mf][nf], false, false);
        accG[mf][nf] = __builtin_amdgcn_wmma_f32_16x16x32_bf16(
            false, ag, false, bfr[nf], (short)0, accG[mf][nf], false, false);
      }
    }
    __syncthreads();
  }

  // ===== epilogue: bias, GLU, residual =====
  const int rowC = (lane >> 4) * 8;
  const int colC = lane & 15;
  #pragma unroll
  for (int mf = 0; mf < 4; ++mf) {
    #pragma unroll
    for (int nf = 0; nf < 2; ++nf) {
      const int lcol = l0 + waveN * 32 + nf * 16 + colC;
      #pragma unroll
      for (int r = 0; r < 8; ++r) {
        const int o = m0 + waveM * 64 + mf * 16 + rowC + r;
        const float val = accV[mf][nf][r] + bias[o];
        const float gat = accG[mf][nf][r] + bias[o + kH];
        const float res = val / (1.f + __expf(-gat));
        const size_t idx = ((size_t)bb * kH + o) * kL + lcol;
        Hres[idx] += res;
      }
    }
  }
}

// ---------------------------------------------------------------------------
// Host launcher
// ---------------------------------------------------------------------------
extern "C" void kernel_launch(void* const* d_in, const int* in_sizes, int n_in,
                              void* d_out, int out_size, void* d_ws, size_t ws_size,
                              hipStream_t stream) {
  (void)in_sizes; (void)n_in; (void)out_size; (void)ws_size;
  const float* x        = (const float*)d_in[0];
  const float* ln_w     = (const float*)d_in[1];
  const float* ln_b     = (const float*)d_in[2];
  const float* log_dt   = (const float*)d_in[3];
  const float* A_re_log = (const float*)d_in[4];
  const float* A_im     = (const float*)d_in[5];
  const float* C_re     = (const float*)d_in[6];
  const float* C_im     = (const float*)d_in[7];
  const float* D_skip   = (const float*)d_in[8];
  const float* W_out    = (const float*)d_in[9];
  const float* b_out    = (const float*)d_in[10];
  float* out = (float*)d_out;

  const size_t nBHL = (size_t)kB * kH * kL;
  float* h_ws = (float*)d_ws;                       // (B,H,L) residual stream
  float* z_ws = h_ws + nBHL;                        // (B,H,L) LN output
  unsigned short* g_ws  = (unsigned short*)(z_ws + nBHL);   // (B,H,L) bf16 act
  unsigned short* wb_ws = g_ws + nBHL;              // (2H,H) bf16 weights
  float* lamr = (float*)(wb_ws + (size_t)2 * kH * kH);
  float* lami = lamr + (size_t)kH * kN;
  float* ccr  = lami + (size_t)kH * kN;
  float* cci  = ccr  + (size_t)kH * kN;

  // x (B,L,H) -> h (B,H,L)
  s4_transpose_in<<<dim3(kL / 32, kH / 32, kB), 256, 0, stream>>>(x, h_ws);

  for (int layer = 0; layer < kNL; ++layer) {
    const size_t oHN = (size_t)layer * kH * kN;
    s4_prep_consts<<<(kH * kN + 255) / 256, 256, 0, stream>>>(
        log_dt + (size_t)layer * kH, A_re_log + oHN, A_im + oHN,
        C_re + oHN, C_im + oHN, lamr, lami, ccr, cci);
    s4_prep_w<<<(2 * kH * kH + 255) / 256, 256, 0, stream>>>(
        W_out + (size_t)layer * 2 * kH * kH, wb_ws, 2 * kH * kH);
    s4_ln_kernel<<<kB * (kL / 32), 256, 0, stream>>>(
        h_ws, ln_w + (size_t)layer * kH, ln_b + (size_t)layer * kH, z_ws);
    s4_scan_kernel<<<(kB * kH) / 8, 256, 0, stream>>>(
        z_ws, lamr, lami, ccr, cci, D_skip + (size_t)layer * kH, g_ws);
    s4_gemm_glu_kernel<<<dim3((kB * kL) / 128, kH / 128), 256, 0, stream>>>(
        wb_ws, g_ws, b_out + (size_t)layer * 2 * kH, h_ws);
  }

  // h (B,H,L) -> out (B,L,H)
  s4_transpose_out<<<dim3(kH / 32, kL / 32, kB), 256, 0, stream>>>(h_ws, out);
}
